// UltraChallengingVQC_21869973471236
// MI455X (gfx1250) — matrix operand order repaired and checked by reading, not compiled, not run on hardware
//
#include <hip/hip_runtime.h>
#include <math.h>

#define QD 256
#define MT 32        // rows per block
#define NW 8         // waves per block
#define TPB 256

typedef __attribute__((ext_vector_type(16))) __bf16          v16bf;
typedef __attribute__((ext_vector_type(8)))  float           v8f;
typedef __attribute__((ext_vector_type(8)))  unsigned short  u16x8;

union AB32 { u16x8 h[2]; v16bf v; };

__device__ __forceinline__ unsigned short f2bf(float f) {
  unsigned u = __float_as_uint(f);
  u += 0x7FFFu + ((u >> 16) & 1u);          // round-to-nearest-even
  return (unsigned short)(u >> 16);
}
__device__ __forceinline__ float bf2f(unsigned short h) {
  return __uint_as_float(((unsigned)h) << 16);
}

__device__ __forceinline__ float wred(float s) {   // wave32 butterfly sum
  s += __shfl_xor(s, 16, 32);
  s += __shfl_xor(s, 8, 32);
  s += __shfl_xor(s, 4, 32);
  s += __shfl_xor(s, 2, 32);
  s += __shfl_xor(s, 1, 32);
  return s;
}

struct Params {
  const float* x;
  float* out;
  const unsigned short* d0w1t[3];
  const unsigned short* d0w2t[3];
  const unsigned short* d1w1t[3];
  const unsigned short* d1w2t[3];
  const unsigned short* wvt[2];
  const unsigned short* wot[2];
  const unsigned short* entt[8];
  const unsigned short* nl1t[4];
  const unsigned short* nl2t[4];
  const float *d0b1, *d0b2, *d1b1, *d1b2, *bqkv, *bo, *nl1b, *nl2b, *lnw, *lnb;
  const float* gatec;   // 8 x 256 x {lin, p3, p4, p5}
};

// D[MT x N] = A[MT x K](bf16 LDS, stride lda) * Wt[N x K](bf16 global, transposed)
// 8 waves split N. Each wave runs a 2x2 grid of 16x16 tiles (two M halves x two
// N tiles) => every B fragment feeds 2 WMMAs (32-row L2 reuse) and 4 chains +
// 4 in-flight b128 loads keep the matrix pipe busy across L2 latency.
__device__ __forceinline__ void gemm_bf16(
    const unsigned short* __restrict__ wt,
    const unsigned short* a, int K, int lda, int N,
    void* out, int ldo, bool outBf16, int wave, int lane)
{
  const int m  = lane & 15;
  const int hi = lane >> 4;
  const unsigned short* ap0 = a + m * lda + hi * 8;        // rows 0..15
  const unsigned short* ap1 = ap0 + 16 * lda;              // rows 16..31
  for (int n0 = wave * 16; n0 < N; n0 += 256) {
    const int n1 = n0 + 128;
    const unsigned short* bp0 = wt + (size_t)(n0 + m) * K + hi * 8;
    const unsigned short* bp1 = wt + (size_t)(n1 + m) * K + hi * 8;
    v8f acc00 = {}; v8f acc10 = {};
    v8f acc01 = {}; v8f acc11 = {};
    for (int k0 = 0; k0 < K; k0 += 32) {
      AB32 av0, av1, bv0, bv1;
      bv0.h[0] = *(const u16x8*)(bp0 + k0);       // col n0+m, K = k0+hi*8..+7, k0+16+hi*8..+7
      bv0.h[1] = *(const u16x8*)(bp0 + k0 + 16);
      bv1.h[0] = *(const u16x8*)(bp1 + k0);
      bv1.h[1] = *(const u16x8*)(bp1 + k0 + 16);
      av0.h[0] = *(const u16x8*)(ap0 + k0);       // K = k0+hi*8   .. +7
      av0.h[1] = *(const u16x8*)(ap0 + k0 + 16);  // K = k0+16+hi*8.. +7
      av1.h[0] = *(const u16x8*)(ap1 + k0);
      av1.h[1] = *(const u16x8*)(ap1 + k0 + 16);
      acc00 = __builtin_amdgcn_wmma_f32_16x16x32_bf16(false, av0.v, false, bv0.v,
                                                      (short)0, acc00, false, false);
      acc10 = __builtin_amdgcn_wmma_f32_16x16x32_bf16(false, av1.v, false, bv0.v,
                                                      (short)0, acc10, false, false);
      acc01 = __builtin_amdgcn_wmma_f32_16x16x32_bf16(false, av0.v, false, bv1.v,
                                                      (short)0, acc01, false, false);
      acc11 = __builtin_amdgcn_wmma_f32_16x16x32_bf16(false, av1.v, false, bv1.v,
                                                      (short)0, acc11, false, false);
    }
    if (outBf16) {
      unsigned short* ob = (unsigned short*)out;
#pragma unroll
      for (int r = 0; r < 8; ++r) {
        ob[(r + 8 * hi) * ldo + n0 + m]      = f2bf(acc00[r]);
        ob[(16 + r + 8 * hi) * ldo + n0 + m] = f2bf(acc10[r]);
        ob[(r + 8 * hi) * ldo + n1 + m]      = f2bf(acc01[r]);
        ob[(16 + r + 8 * hi) * ldo + n1 + m] = f2bf(acc11[r]);
      }
    } else {
      float* of = (float*)out;
#pragma unroll
      for (int r = 0; r < 8; ++r) {
        of[(r + 8 * hi) * ldo + n0 + m]      = acc00[r];
        of[(16 + r + 8 * hi) * ldo + n0 + m] = acc10[r];
        of[(r + 8 * hi) * ldo + n1 + m]      = acc01[r];
        of[(16 + r + 8 * hi) * ldo + n1 + m] = acc11[r];
      }
    }
  }
}

__launch_bounds__(TPB, 4)
__global__ void vqc_fused(Params p)
{
  __shared__ float curf[MT * QD];                          // fp32 current activation (32 KB)
  __shared__ __align__(16) unsigned short abuf[MT * 776];  // bf16 A operand, padded rows
  __shared__ __align__(16) unsigned short obuf[MT * 776];  // GEMM out: bf16 (wide) / fp32 (narrow, aliased)
  float* const ofl = (float*)obuf;                         // fp32 view, 32x256 fits in-place

  const int tid  = threadIdx.x;
  const int lane = tid & 31, wave = tid >> 5;
  const size_t row0 = (size_t)blockIdx.x * MT;
  const float* xin = p.x + row0 * QD;

#pragma unroll
  for (int i = 0; i < 32; ++i) {
    int idx = tid + i * TPB;
    float v = xin[idx];
    curf[idx] = v;
    abuf[(idx >> 8) * 264 + (idx & 255)] = f2bf(v);
  }
  __syncthreads();

  float res[32];

#pragma unroll 1
  for (int li = 0; li < 8; ++li) {
#pragma unroll
    for (int i = 0; i < 32; ++i) res[i] = curf[tid + i * TPB];
    const int t = li % 3;

    if (t == 0) {
      const int iw = li / 3;
      gemm_bf16(p.d0w1t[iw], abuf, QD, 264, 768, obuf, 768, true, wave, lane);
      __syncthreads();
      const float* b1 = p.d0b1 + iw * 768;
      const float* lw = p.lnw + iw * 768;
      const float* lb = p.lnb + iw * 768;
      for (int m = wave; m < MT; m += NW) {
        float s = 0.f, q2 = 0.f;
        for (int e = lane; e < 768; e += 32) {
          float v = bf2f(obuf[m * 768 + e]) + b1[e];
          s += v; q2 += v * v;
        }
        s = wred(s); q2 = wred(q2);
        float mu  = s * (1.f / 768.f);
        float var = q2 * (1.f / 768.f) - mu * mu;
        float rin = rsqrtf(var + 1e-5f);
        for (int e = lane; e < 768; e += 32) {
          float v = bf2f(obuf[m * 768 + e]) + b1[e];
          v = (v - mu) * rin * lw[e] + lb[e];
          v = 0.5f * v * (1.f + erff(v * 0.70710678118654752f));  // exact gelu
          abuf[m * 776 + e] = f2bf(v);
        }
      }
      __syncthreads();
      gemm_bf16(p.d0w2t[iw], abuf, 768, 776, QD, obuf, QD, false, wave, lane);
      __syncthreads();
      const float* b2 = p.d0b2 + iw * QD;
#pragma unroll
      for (int i = 0; i < 32; ++i) {
        int idx = tid + i * TPB;
        float v = ofl[idx] + b2[idx & 255];
        curf[idx] = v;
        abuf[(idx >> 8) * 264 + (idx & 255)] = f2bf(v);
      }
      __syncthreads();
    } else if (t == 1) {
      const int iw = (li - 1) / 3;
      gemm_bf16(p.d1w1t[iw], abuf, QD, 264, 512, obuf, 512, true, wave, lane);
      __syncthreads();
      const float* b1 = p.d1b1 + iw * 512;
      for (int m = wave; m < MT; m += NW)
        for (int e = lane; e < 512; e += 32) {
          float v = bf2f(obuf[m * 512 + e]) + b1[e];
          v = v / (1.f + expf(-v));                 // silu
          abuf[m * 520 + e] = f2bf(v);
        }
      __syncthreads();
      gemm_bf16(p.d1w2t[iw], abuf, 512, 520, QD, obuf, QD, false, wave, lane);
      __syncthreads();
      const float* b2 = p.d1b2 + iw * QD;
#pragma unroll
      for (int i = 0; i < 32; ++i) {
        int idx = tid + i * TPB;
        float v = ofl[idx] + b2[idx & 255];
        curf[idx] = v;
        abuf[(idx >> 8) * 264 + (idx & 255)] = f2bf(v);
      }
      __syncthreads();
    } else {
      const int iw = (li - 2) / 3;
      // only the V slice of QKV is ever used
      gemm_bf16(p.wvt[iw], abuf, QD, 264, QD, obuf, QD, false, wave, lane);
      __syncthreads();
      const float* bv = p.bqkv + iw * 768 + 512;
#pragma unroll
      for (int i = 0; i < 32; ++i) {
        int idx = tid + i * TPB;
        float v = ofl[idx] + bv[idx & 255];
        abuf[(idx >> 8) * 264 + (idx & 255)] = f2bf(v);
      }
      __syncthreads();
      gemm_bf16(p.wot[iw], abuf, QD, 264, QD, obuf, QD, false, wave, lane);
      __syncthreads();
      const float* b2 = p.bo + iw * QD;
#pragma unroll
      for (int i = 0; i < 32; ++i) {
        int idx = tid + i * TPB;
        float v = ofl[idx] + b2[idx & 255];
        curf[idx] = v;
        abuf[(idx >> 8) * 264 + (idx & 255)] = f2bf(v);
      }
      __syncthreads();
    }

    // entangled = cur @ tanh(ent[li])   (tanh folded into weights at prep)
    gemm_bf16(p.entt[li], abuf, QD, 264, QD, obuf, QD, false, wave, lane);
    __syncthreads();

    // gate + mix
    const float* gc = p.gatec + (size_t)li * QD * 4;
#pragma unroll
    for (int i = 0; i < 32; ++i) {
      int idx = tid + i * TPB;
      int q = idx & 255;
      float c   = curf[idx];
      float lin = gc[q * 4 + 0], p3 = gc[q * 4 + 1];
      float p4  = gc[q * 4 + 2], p5 = gc[q * 4 + 3];
      float gate = (lin * c + 0.5f * sinf(p3 * c + p4) + 0.5f * cosf(p5 * c)) * 0.25f;
      c = (c + 0.3f * gate + 0.2f * ofl[idx]) * (1.0f / 1.5f);
      curf[idx] = c;
      if (li & 1) abuf[(idx >> 8) * 264 + q] = f2bf(c);
    }
    __syncthreads();

    if (li & 1) {
      const int j = li / 2;
      gemm_bf16(p.nl1t[j], abuf, QD, 264, QD, obuf, QD, false, wave, lane);
      __syncthreads();
      const float* b1 = p.nl1b + j * QD;
#pragma unroll
      for (int i = 0; i < 32; ++i) {
        int idx = tid + i * TPB;
        float v = tanhf(ofl[idx] + b1[idx & 255]);
        abuf[(idx >> 8) * 264 + (idx & 255)] = f2bf(v);
      }
      __syncthreads();
      gemm_bf16(p.nl2t[j], abuf, QD, 264, QD, obuf, QD, false, wave, lane);
      __syncthreads();
      const float* b2 = p.nl2b + j * QD;
#pragma unroll
      for (int i = 0; i < 32; ++i) {
        int idx = tid + i * TPB;
        curf[idx] += 0.1f * (ofl[idx] + b2[idx & 255]);
      }
      __syncthreads();
    }

    // residual blend, row L2-normalize, tanh; prepack next A
    const float alpha = (li < 4) ? 0.8f : 0.6f;
#pragma unroll
    for (int i = 0; i < 32; ++i) {
      int idx = tid + i * TPB;
      curf[idx] = alpha * curf[idx] + (1.f - alpha) * res[i];
    }
    __syncthreads();
    for (int m = wave; m < MT; m += NW) {
      float s = 0.f;
      for (int e = lane; e < QD; e += 32) { float v = curf[m * QD + e]; s += v * v; }
      s = wred(s);
      float inv = 1.f / (sqrtf(s) + 1e-8f);
      for (int e = lane; e < QD; e += 32) {
        float v = tanhf(curf[m * QD + e] * inv);
        curf[m * QD + e] = v;
        abuf[m * 264 + e] = f2bf(v);
      }
    }
    __syncthreads();
  }

  float* o = p.out + row0 * QD;
#pragma unroll
  for (int i = 0; i < 32; ++i) {
    int idx = tid + i * TPB;
    o[idx] = curf[idx];
  }
}

// dst[n*K+k] = bf16( maybe_tanh( src[k*srcLd + colOff + n] ) )   (transpose to N x K)
__global__ void prep_mat(const float* __restrict__ src, unsigned short* __restrict__ dst,
                         int K, int N, int srcLd, int colOff, int doTanh)
{
  int idx = blockIdx.x * 256 + threadIdx.x;
  if (idx >= K * N) return;
  int n = idx / K, k = idx - n * K;
  float v = src[(size_t)k * srcLd + colOff + n];
  if (doTanh) v = tanhf(v);
  dst[idx] = f2bf(v);
}

__global__ void prep_gate(const float* __restrict__ gp, float* __restrict__ gc)
{
  int i = blockIdx.x * 256 + threadIdx.x;
  if (i >= 8 * 256) return;
  const float* q = gp + (size_t)i * 6;
  gc[i * 4 + 0] = sinf(q[0]) + cosf(q[1]) + tanhf(q[2]);
  gc[i * 4 + 1] = q[3];
  gc[i * 4 + 2] = q[4];
  gc[i * 4 + 3] = q[5];
}

extern "C" void kernel_launch(void* const* d_in, const int* in_sizes, int n_in,
                              void* d_out, int out_size, void* d_ws, size_t ws_size,
                              hipStream_t stream)
{
  const float* x    = (const float*)d_in[0];
  const float* d0w1 = (const float*)d_in[1];
  const float* d0b1 = (const float*)d_in[2];
  const float* lnw  = (const float*)d_in[3];
  const float* lnb  = (const float*)d_in[4];
  const float* d0w2 = (const float*)d_in[5];
  const float* d0b2 = (const float*)d_in[6];
  const float* d1w1 = (const float*)d_in[7];
  const float* d1b1 = (const float*)d_in[8];
  const float* d1w2 = (const float*)d_in[9];
  const float* d1b2 = (const float*)d_in[10];
  const float* wqkv = (const float*)d_in[11];
  const float* bqkv = (const float*)d_in[12];
  const float* wo   = (const float*)d_in[13];
  const float* bo   = (const float*)d_in[14];
  const float* gp   = (const float*)d_in[15];
  const float* ent  = (const float*)d_in[16];
  const float* nl1w = (const float*)d_in[17];
  const float* nl1b = (const float*)d_in[18];
  const float* nl2w = (const float*)d_in[19];
  const float* nl2b = (const float*)d_in[20];

  unsigned char* ws = (unsigned char*)d_ws;
  size_t off = 0;
  auto alloc = [&](size_t bytes) -> void* {
    off = (off + 255) & ~(size_t)255;
    void* p = ws + off;
    off += bytes;
    return p;
  };

  Params P;
  P.x = x; P.out = (float*)d_out;
  P.d0b1 = d0b1; P.d0b2 = d0b2; P.d1b1 = d1b1; P.d1b2 = d1b2;
  P.bqkv = bqkv; P.bo = bo; P.nl1b = nl1b; P.nl2b = nl2b;
  P.lnw = lnw;  P.lnb = lnb;

  auto prep = [&](const float* src, int K, int N, int srcLd, int colOff,
                  int doTanh) -> unsigned short* {
    unsigned short* dst = (unsigned short*)alloc((size_t)K * N * 2);
    int total = K * N;
    prep_mat<<<(total + 255) / 256, 256, 0, stream>>>(src, dst, K, N, srcLd, colOff, doTanh);
    return dst;
  };

  for (int i = 0; i < 3; ++i) P.d0w1t[i] = prep(d0w1 + (size_t)i * 256 * 768, 256, 768, 768, 0, 0);
  for (int i = 0; i < 3; ++i) P.d0w2t[i] = prep(d0w2 + (size_t)i * 768 * 256, 768, 256, 256, 0, 0);
  for (int i = 0; i < 3; ++i) P.d1w1t[i] = prep(d1w1 + (size_t)i * 256 * 512, 256, 512, 512, 0, 0);
  for (int i = 0; i < 3; ++i) P.d1w2t[i] = prep(d1w2 + (size_t)i * 512 * 256, 512, 256, 256, 0, 0);
  for (int i = 0; i < 2; ++i) P.wvt[i]  = prep(wqkv + (size_t)i * 256 * 768, 256, 256, 768, 512, 0);
  for (int i = 0; i < 2; ++i) P.wot[i]  = prep(wo   + (size_t)i * 256 * 256, 256, 256, 256, 0, 0);
  for (int i = 0; i < 8; ++i) P.entt[i] = prep(ent  + (size_t)i * 256 * 256, 256, 256, 256, 0, 1);
  for (int i = 0; i < 4; ++i) P.nl1t[i] = prep(nl1w + (size_t)i * 256 * 256, 256, 256, 256, 0, 0);
  for (int i = 0; i < 4; ++i) P.nl2t[i] = prep(nl2w + (size_t)i * 256 * 256, 256, 256, 256, 0, 0);

  float* gc = (float*)alloc(8 * 256 * 4 * sizeof(float));
  P.gatec = gc;
  prep_gate<<<8, 256, 0, stream>>>(gp, gc);

  vqc_fused<<<65536 / MT, TPB, 0, stream>>>(P);
}
